// Transformer_8091718385730
// MI455X (gfx1250) — matrix operand order repaired and checked by reading, not compiled, not run on hardware
//
#include <hip/hip_runtime.h>
#include <hip/hip_bf16.h>
#include <math.h>

// ---------------- problem constants (match reference) ----------------
#define N_TOK 2048
#define DIM   2048
#define NH    16
#define NOPE  128
#define ROPE  64
#define VH    128
#define QKH   192          // NOPE + ROPE
#define KVR   512
#define INTER 5632
#define VOCAB 32000
#define NLAYER 2
#define KVD   (NOPE + VH)  // 256

typedef __attribute__((ext_vector_type(16))) __bf16 v16bf;
typedef __attribute__((ext_vector_type(8)))  __bf16 v8bf;
typedef __attribute__((ext_vector_type(4)))  __bf16 v4bf;
typedef __attribute__((ext_vector_type(8)))  float  v8f;
typedef __attribute__((ext_vector_type(4)))  float  v4f;

// WMMA fragment layouts as loaded here, wave32:
//  A 16x32 bf16 : lane%16 = row M; khalf = lane/16; elements = one contiguous
//                 K-run [khalf*16,+16), as B
//  B 32x16 bf16 : lane%16 = col N; elements = contiguous K-run [khalf*16,+16)
//  C/D 16x16 f32: lane%16 = col N; VGPR i -> row M = i + 8*khalf

__device__ __forceinline__ v16bf ld8x2(const __bf16* p0, const __bf16* p1) {
  v8bf a = *(const v8bf*)p0;
  v8bf b = *(const v8bf*)p1;
  return __builtin_shufflevector(a, b, 0,1,2,3,4,5,6,7,8,9,10,11,12,13,14,15);
}
__device__ __forceinline__ v16bf ld16(const __bf16* p) {
  return *(const v16bf*)p;   // 32B-aligned contiguous run -> 2x b128
}
__device__ __forceinline__ v8f wmma_bf16(v16bf a, v16bf b, v8f c) {
  return __builtin_amdgcn_wmma_f32_16x16x32_bf16(false, a, false, b,
                                                 (short)0, c, false, false);
}

__device__ __forceinline__ float half16_max(float v) {
#pragma unroll
  for (int m = 1; m < 16; m <<= 1) v = fmaxf(v, __shfl_xor(v, m, 32));
  return v;
}
__device__ __forceinline__ float half16_sum(float v) {
#pragma unroll
  for (int m = 1; m < 16; m <<= 1) v += __shfl_xor(v, m, 32);
  return v;
}
__device__ __forceinline__ int imin(int a, int b) { return a < b ? a : b; }

// ---------------- embedding gather ----------------
__global__ void embed_kernel(const int* __restrict__ tokens,
                             const float* __restrict__ emb,
                             float* __restrict__ h) {
  int n = blockIdx.x;
  int t = tokens[n];
  for (int i = threadIdx.x; i < DIM; i += blockDim.x)
    h[(unsigned)n * DIM + i] = emb[(unsigned)t * DIM + i];
}

// ---------------- strided RMSNorm ----------------
__global__ __launch_bounds__(256)
void rmsnorm_kernel(const float* __restrict__ in, const float* __restrict__ w,
                    float* __restrict__ out, int width, int in_stride, int in_ofs,
                    int out_stride) {
  int row = blockIdx.x;
  const float* x = in + (unsigned)row * in_stride + in_ofs;
  float* y = out + (unsigned)row * out_stride;
  __shared__ float red[8];
  float ss = 0.f;
  for (int i = threadIdx.x; i < width; i += 256) { float v = x[i]; ss += v * v; }
#pragma unroll
  for (int m = 1; m < 32; m <<= 1) ss += __shfl_xor(ss, m, 32);
  if ((threadIdx.x & 31) == 0) red[threadIdx.x >> 5] = ss;
  __syncthreads();
  float tot = red[0] + red[1] + red[2] + red[3] + red[4] + red[5] + red[6] + red[7];
  float r = rsqrtf(tot / (float)width + 1e-6f);
  for (int i = threadIdx.x; i < width; i += 256) y[i] = x[i] * r * w[i];
}

// ---------------- RoPE ----------------
__global__ __launch_bounds__(32)
void rope_q_kernel(float* __restrict__ q, const float* __restrict__ cs,
                   const float* __restrict__ sn) {
  int head = blockIdx.x, n = blockIdx.y, d = threadIdx.x;   // d in 0..31
  float c = cs[(unsigned)n * (ROPE / 2) + d];
  float s = sn[(unsigned)n * (ROPE / 2) + d];
  float* p = q + ((unsigned)n * NH + head) * QKH + NOPE + 2 * d;
  float xr = p[0], xi = p[1];
  p[0] = xr * c - xi * s;
  p[1] = xr * s + xi * c;
}

__global__ __launch_bounds__(32)
void rope_k_kernel(const float* __restrict__ kv, float* __restrict__ kr,
                   const float* __restrict__ cs, const float* __restrict__ sn) {
  int n = blockIdx.x, d = threadIdx.x;
  float c = cs[(unsigned)n * (ROPE / 2) + d];
  float s = sn[(unsigned)n * (ROPE / 2) + d];
  const float* p = kv + (unsigned)n * (ROPE + KVR) + 2 * d;
  float xr = p[0], xi = p[1];
  kr[(unsigned)n * ROPE + 2 * d]     = xr * c - xi * s;
  kr[(unsigned)n * ROPE + 2 * d + 1] = xr * s + xi * c;
}

// ---------------- SiLU gate ----------------
__global__ void silu_mul_kernel(float* __restrict__ t1, const float* __restrict__ t2,
                                unsigned count) {
  unsigned i = blockIdx.x * blockDim.x + threadIdx.x;
  if (i < count) {
    float a = t1[i];
    t1[i] = (a / (1.0f + __expf(-a))) * t2[i];
  }
}

// ---------------- attention input packing ----------------
__global__ __launch_bounds__(256)
void pack_k_kernel(const float* __restrict__ kvb, const float* __restrict__ kr,
                   __bf16* __restrict__ kb16) {
  int n = blockIdx.x;
  for (int i = threadIdx.x; i < NH * QKH; i += 256) {
    int head = i / QKH, d = i - head * QKH;
    float v = (d < NOPE) ? kvb[((unsigned)n * NH + head) * KVD + d]
                         : kr[(unsigned)n * ROPE + (d - NOPE)];
    kb16[((unsigned)head * N_TOK + n) * QKH + d] = (__bf16)v;
  }
}

__global__ __launch_bounds__(256)
void pack_v_kernel(const float* __restrict__ kvb, __bf16* __restrict__ vT) {
  __shared__ float tile[32][33];
  int head = blockIdx.z;
  int n0 = blockIdx.x * 32;
  int v0 = blockIdx.y * 32;
  int tx = threadIdx.x & 31;
  int ty = threadIdx.x >> 5;
#pragma unroll
  for (int it = 0; it < 4; ++it) {
    int n = n0 + ty + it * 8;
    tile[ty + it * 8][tx] = kvb[((unsigned)n * NH + head) * KVD + NOPE + v0 + tx];
  }
  __syncthreads();
#pragma unroll
  for (int it = 0; it < 4; ++it) {
    int vd = v0 + ty + it * 8;
    vT[(unsigned)(head * VH + vd) * N_TOK + n0 + tx] = (__bf16)tile[tx][ty + it * 8];
  }
}

// ---------------- bf16-WMMA GEMM, 128 x TBN tile, double-buffered LDS ----------------
#define BM 128
#define BK 32
#define LDA 48   // 96B rows: 16/32B-aligned fragment runs, bank-spread
#define LDB 48
template <int TBN>
__global__ __launch_bounds__(256)
void gemm_wmma(const float* __restrict__ A, const float* __restrict__ B,
               float* __restrict__ C, unsigned M, unsigned Nn, unsigned K, int mode) {
  constexpr int NCH   = TBN / 64;       // 32-col half-pairs per wave
  constexpr int KROWS = 1024 / TBN;     // B k-rows staged per pass
  constexpr int BITS  = BK / KROWS;     // B staging iterations
  __shared__ __bf16 As[2][BM * LDA];    // 2 x 12 KB
  __shared__ __bf16 Bt[2][TBN * LDB];   // 2 x (6|12) KB, transposed [n][k]

  unsigned row0 = blockIdx.y * BM;
  unsigned col0 = blockIdx.x * TBN;
  unsigned tid  = threadIdx.x;
  unsigned wave = tid >> 5, lane = tid & 31;
  unsigned wrow = (wave >> 1) * 32;           // 4 row-strips of 32
  unsigned wcol = (wave & 1) * (TBN / 2);     // 2 col-strips
  unsigned lrow = lane & 15, khalf = lane >> 4;

  unsigned arow = tid >> 3;                   // 0..31 (+32/it)
  unsigned acol = (tid & 7) * 4;
  unsigned brow = tid / (TBN / 4);            // k index
  unsigned bcol = (tid % (TBN / 4)) * 4;      // n index

  const float* aP = A + (row0 + arow) * K + acol;
  const float* bP = B + brow * Nn + col0 + bcol;
  const unsigned bstep = BK * Nn;
  __bf16* asd_base = &As[0][arow * LDA + acol];
  __bf16* bsd_base = &Bt[0][bcol * LDB + brow];
  constexpr unsigned ABUF = BM * LDA;
  constexpr unsigned BBUF = (unsigned)TBN * LDB;

  v4f ra[4], rb[BITS];
#pragma unroll
  for (int it = 0; it < 4; ++it) ra[it] = *(const v4f*)(aP + it * 32u * K);
#pragma unroll
  for (int it = 0; it < BITS; ++it) rb[it] = *(const v4f*)(bP + it * KROWS * Nn);

  v8f acc[2][2 * NCH];
#pragma unroll
  for (int rh = 0; rh < 2; ++rh)
#pragma unroll
    for (int c = 0; c < 2 * NCH; ++c) acc[rh][c] = (v8f){};

  unsigned buf = 0;
  {
#pragma unroll
    for (int it = 0; it < 4; ++it) {
      v4bf p;
#pragma unroll
      for (int j = 0; j < 4; ++j) p[j] = (__bf16)ra[it][j];
      *(v4bf*)(asd_base + it * 32 * LDA) = p;
    }
#pragma unroll
    for (int it = 0; it < BITS; ++it)
#pragma unroll
      for (int j = 0; j < 4; ++j)
        bsd_base[(unsigned)j * LDB + it * KROWS] = (__bf16)rb[it][j];
  }
  __syncthreads();

  for (unsigned k0 = BK; k0 < K; k0 += BK) {
    aP += BK; bP += bstep;
#pragma unroll
    for (int it = 0; it < 4; ++it) ra[it] = *(const v4f*)(aP + it * 32u * K);
#pragma unroll
    for (int it = 0; it < BITS; ++it) rb[it] = *(const v4f*)(bP + it * KROWS * Nn);

    {
      const __bf16* ab = &As[buf][(wrow + lrow) * LDA + khalf * 16];
      v16bf af0 = ld8x2(ab, ab + 8);
      v16bf af1 = ld8x2(ab + 16 * LDA, ab + 16 * LDA + 8);
#pragma unroll
      for (int ch = 0; ch < NCH; ++ch) {
        const __bf16* bb = &Bt[buf][(wcol + ch * 32 + lrow) * LDB + khalf * 16];
        v16bf bf0 = ld16(bb);
        v16bf bf1 = ld16(bb + 16 * LDB);
        acc[0][2 * ch + 0] = wmma_bf16(af0, bf0, acc[0][2 * ch + 0]);
        acc[0][2 * ch + 1] = wmma_bf16(af0, bf1, acc[0][2 * ch + 1]);
        acc[1][2 * ch + 0] = wmma_bf16(af1, bf0, acc[1][2 * ch + 0]);
        acc[1][2 * ch + 1] = wmma_bf16(af1, bf1, acc[1][2 * ch + 1]);
      }
    }
    {
      __bf16* asd = asd_base + (buf ^ 1) * ABUF;
      __bf16* bsd = bsd_base + (buf ^ 1) * BBUF;
#pragma unroll
      for (int it = 0; it < 4; ++it) {
        v4bf p;
#pragma unroll
        for (int j = 0; j < 4; ++j) p[j] = (__bf16)ra[it][j];
        *(v4bf*)(asd + it * 32 * LDA) = p;
      }
#pragma unroll
      for (int it = 0; it < BITS; ++it)
#pragma unroll
        for (int j = 0; j < 4; ++j)
          bsd[(unsigned)j * LDB + it * KROWS] = (__bf16)rb[it][j];
    }
    __syncthreads();
    buf ^= 1;
  }

  {
    const __bf16* ab = &As[buf][(wrow + lrow) * LDA + khalf * 16];
    v16bf af0 = ld8x2(ab, ab + 8);
    v16bf af1 = ld8x2(ab + 16 * LDA, ab + 16 * LDA + 8);
#pragma unroll
    for (int ch = 0; ch < NCH; ++ch) {
      const __bf16* bb = &Bt[buf][(wcol + ch * 32 + lrow) * LDB + khalf * 16];
      v16bf bf0 = ld16(bb);
      v16bf bf1 = ld16(bb + 16 * LDB);
      acc[0][2 * ch + 0] = wmma_bf16(af0, bf0, acc[0][2 * ch + 0]);
      acc[0][2 * ch + 1] = wmma_bf16(af0, bf1, acc[0][2 * ch + 1]);
      acc[1][2 * ch + 0] = wmma_bf16(af1, bf0, acc[1][2 * ch + 0]);
      acc[1][2 * ch + 1] = wmma_bf16(af1, bf1, acc[1][2 * ch + 1]);
    }
  }

#pragma unroll
  for (int rh = 0; rh < 2; ++rh) {
#pragma unroll
    for (int ch = 0; ch < NCH; ++ch) {
      unsigned cc = col0 + wcol + ch * 32 + lrow;
#pragma unroll
      for (int i = 0; i < 8; ++i) {
        unsigned r = row0 + wrow + rh * 16 + i + 8 * khalf;
        float* p0 = &C[r * Nn + cc];
        float* p1 = p0 + 16;
        if (mode == 1) { *p0 += acc[rh][2 * ch][i]; *p1 += acc[rh][2 * ch + 1][i]; }
        else           { *p0  = acc[rh][2 * ch][i]; *p1  = acc[rh][2 * ch + 1][i]; }
      }
    }
  }
}

// ---------------- flash attention (causal, MLA) ----------------
__global__ __launch_bounds__(32)
void attn_wmma(const float* __restrict__ q, const __bf16* __restrict__ kb16,
               const __bf16* __restrict__ vT, float* __restrict__ o, int n) {
  __shared__ __bf16 Pl[16 * 32];
  int qbase = blockIdx.x * 16;
  int head  = blockIdx.y;
  int lane  = threadIdx.x & 31;
  int lrow  = lane & 15;
  int khalf = lane >> 4;

  v16bf qf[6];
  {
    const float* qp = q + ((unsigned)(qbase + lrow) * NH + head) * QKH + khalf * 16;
#pragma unroll
    for (int s = 0; s < 6; ++s) {
      const float* p = qp + s * 32;
      v4f a = *(const v4f*)p;
      v4f b = *(const v4f*)(p + 4);
      v4f c = *(const v4f*)(p + 8);
      v4f d = *(const v4f*)(p + 12);
#pragma unroll
      for (int j = 0; j < 4; ++j) {
        qf[s][j]      = (__bf16)a[j];
        qf[s][4 + j]  = (__bf16)b[j];
        qf[s][8 + j]  = (__bf16)c[j];
        qf[s][12 + j] = (__bf16)d[j];
      }
    }
  }

  v8f oacc[8];
#pragma unroll
  for (int t = 0; t < 8; ++t) oacc[t] = (v8f){};
  float mrow[8], lsum[8];
#pragma unroll
  for (int i = 0; i < 8; ++i) { mrow[i] = -3.0e38f; lsum[i] = 0.f; }

  const float scale = 0.07216878364870323f;   // 1/sqrt(192)
  int nkt = (qbase + 16 + 31) >> 5;

  const __bf16* kbase_p = kb16 + (unsigned)head * N_TOK * QKH;
  const __bf16* vbase_p = vT + (unsigned)head * VH * N_TOK;

  for (int kt = 0; kt < nkt; ++kt) {
    int kb = kt * 32;
    if (kt + 1 < nkt) {   // gfx1250 global_prefetch_b8 for next key tile
      int pk = imin(kb + 32 + lane, n - 1);
      __builtin_prefetch(kbase_p + (unsigned)pk * QKH, 0, 1);
      __builtin_prefetch(vbase_p + (unsigned)(lane * 4) * N_TOK + kb + 32, 0, 1);
    }

    // ---- S = Q K^T : 12-step 2-deep software pipeline (distinct regs so the
    // wait before each WMMA leaves the next fragment's loads in flight) ----
    int keyc0 = imin(kb + lrow, n - 1);
    int keyc1 = imin(kb + 16 + lrow, n - 1);
    const __bf16* kp0 = kbase_p + (unsigned)keyc0 * QKH + khalf * 16;
    const __bf16* kp1 = kbase_p + (unsigned)keyc1 * QKH + khalf * 16;
    v8f s0 = {}, s1 = {};
    {
      v16bf cur = ld16(kp0);
#pragma unroll
      for (int u = 0; u < 12; ++u) {
        v16bf nxt = cur;
        if (u + 1 < 12) {
          int un = u + 1;
          const __bf16* p = (un < 6) ? kp0 : kp1;
          nxt = ld16(p + (un % 6) * 32);
        }
        if (u < 6) s0 = wmma_bf16(qf[u], cur, s0);
        else       s1 = wmma_bf16(qf[u - 6], cur, s1);
        cur = nxt;
      }
    }

    // causal mask + online softmax; row = i + 8*khalf lives in one 16-lane half
    int kkey0 = kb + lrow;
    int kkey1 = kb + 16 + lrow;
#pragma unroll
    for (int i = 0; i < 8; ++i) {
      int r = qbase + i + 8 * khalf;
      float v0 = s0[i] * scale; if (kkey0 > r) v0 = -1.0e30f;
      float v1 = s1[i] * scale; if (kkey1 > r) v1 = -1.0e30f;
      float tmax = half16_max(fmaxf(v0, v1));
      float mnew = fmaxf(mrow[i], tmax);
      float alpha = __expf(mrow[i] - mnew);
      float p0 = __expf(v0 - mnew);
      float p1 = __expf(v1 - mnew);
      lsum[i] = lsum[i] * alpha + half16_sum(p0 + p1);
      mrow[i] = mnew;
#pragma unroll
      for (int t = 0; t < 8; ++t) oacc[t][i] *= alpha;
      Pl[(i + 8 * khalf) * 32 + lrow]      = (__bf16)p0;
      Pl[(i + 8 * khalf) * 32 + 16 + lrow] = (__bf16)p1;
    }
    __syncthreads();

    // ---- O += P V : 8-step 2-deep pipeline over value chunks ----
    const __bf16* pp = &Pl[lrow * 32 + khalf * 16];
    v16bf pf = ld8x2(pp, pp + 8);
    {
      const __bf16* vp = vbase_p + (unsigned)lrow * N_TOK + kb + khalf * 16;
      v16bf vcur = ld16(vp);
#pragma unroll
      for (int t = 0; t < 8; ++t) {
        v16bf vnxt = vcur;
        if (t + 1 < 8) vnxt = ld16(vp + (unsigned)(t + 1) * 16 * N_TOK);
        oacc[t] = wmma_bf16(pf, vcur, oacc[t]);
        vcur = vnxt;
      }
    }
    __syncthreads();
  }

#pragma unroll
  for (int i = 0; i < 8; ++i) {
    float inv = 1.0f / lsum[i];
    unsigned r = qbase + i + 8 * khalf;
#pragma unroll
    for (int t = 0; t < 8; ++t)
      o[(r * NH + head) * VH + t * 16 + lrow] = oacc[t][i] * inv;
  }
}

// ---------------- host orchestration ----------------
static inline void run_gemm(const float* A, const float* B, float* C,
                            unsigned M, unsigned Nn, unsigned K, int mode,
                            hipStream_t s) {
  if (Nn % 128 == 0) {
    dim3 g(Nn / 128, M / BM);
    gemm_wmma<128><<<g, 256, 0, s>>>(A, B, C, M, Nn, K, mode);
  } else {
    dim3 g(Nn / 64, M / BM);
    gemm_wmma<64><<<g, 256, 0, s>>>(A, B, C, M, Nn, K, mode);
  }
}

extern "C" void kernel_launch(void* const* d_in, const int* in_sizes, int n_in,
                              void* d_out, int out_size, void* d_ws, size_t ws_size,
                              hipStream_t stream) {
  (void)in_sizes; (void)n_in; (void)out_size; (void)ws_size;

  const int*   tokens   = (const int*)d_in[0];
  const float* emb      = (const float*)d_in[1];
  const float* qW       = (const float*)d_in[2];
  const float* kvaW     = (const float*)d_in[3];
  const float* kvnorm_w = (const float*)d_in[4];
  const float* kvbW     = (const float*)d_in[5];
  const float* oW       = (const float*)d_in[6];
  const float* w1       = (const float*)d_in[7];
  const float* w2       = (const float*)d_in[8];
  const float* w3       = (const float*)d_in[9];
  const float* attn_nw  = (const float*)d_in[10];
  const float* ffn_nw   = (const float*)d_in[11];
  const float* norm_w   = (const float*)d_in[12];
  const float* outW     = (const float*)d_in[13];
  const float* rope_cos = (const float*)d_in[14];
  const float* rope_sin = (const float*)d_in[15];
  float* logits = (float*)d_out;

  float* ws = (float*)d_ws;
  size_t ofs = 0;
  float* h   = ws + ofs; ofs += (size_t)N_TOK * DIM;
  float* x   = ws + ofs; ofs += (size_t)N_TOK * DIM;
  float* qb  = ws + ofs; ofs += (size_t)N_TOK * NH * QKH;
  float* kv  = ws + ofs; ofs += (size_t)N_TOK * (ROPE + KVR);
  float* cb  = ws + ofs; ofs += (size_t)N_TOK * KVR;
  float* kvb = ws + ofs; ofs += (size_t)N_TOK * NH * KVD;
  float* kr  = ws + ofs; ofs += (size_t)N_TOK * ROPE;
  float* ob  = ws + ofs; ofs += (size_t)N_TOK * NH * VH;
  float* t1  = ws + ofs; ofs += (size_t)N_TOK * INTER;
  float* t2  = ws + ofs; ofs += (size_t)N_TOK * INTER;
  __bf16* kb16 = (__bf16*)(ws + ofs); ofs += ((size_t)NH * N_TOK * QKH + 64) / 2;
  __bf16* vT   = (__bf16*)(ws + ofs); ofs += ((size_t)NH * VH * N_TOK + 64) / 2;

  embed_kernel<<<N_TOK, 256, 0, stream>>>(tokens, emb, h);

  for (int l = 0; l < NLAYER; ++l) {
    const float* qW_l   = qW   + (size_t)l * DIM * NH * QKH;
    const float* kvaW_l = kvaW + (size_t)l * DIM * (ROPE + KVR);
    const float* kvbW_l = kvbW + (size_t)l * KVR * NH * KVD;
    const float* oW_l   = oW   + (size_t)l * NH * VH * DIM;
    const float* w1_l   = w1   + (size_t)l * DIM * INTER;
    const float* w2_l   = w2   + (size_t)l * DIM * INTER;
    const float* w3_l   = w3   + (size_t)l * INTER * DIM;

    rmsnorm_kernel<<<N_TOK, 256, 0, stream>>>(h, attn_nw + (size_t)l * DIM, x,
                                              DIM, DIM, 0, DIM);
    run_gemm(x, qW_l,   qb, N_TOK, NH * QKH,   DIM, 0, stream);
    run_gemm(x, kvaW_l, kv, N_TOK, ROPE + KVR, DIM, 0, stream);
    rope_q_kernel<<<dim3(NH, N_TOK), 32, 0, stream>>>(qb, rope_cos, rope_sin);
    rope_k_kernel<<<N_TOK, 32, 0, stream>>>(kv, kr, rope_cos, rope_sin);
    rmsnorm_kernel<<<N_TOK, 256, 0, stream>>>(kv, kvnorm_w + (size_t)l * KVR, cb,
                                              KVR, ROPE + KVR, ROPE, KVR);
    run_gemm(cb, kvbW_l, kvb, N_TOK, NH * KVD, KVR, 0, stream);
    pack_k_kernel<<<N_TOK, 256, 0, stream>>>(kvb, kr, kb16);
    pack_v_kernel<<<dim3(N_TOK / 32, VH / 32, NH), 256, 0, stream>>>(kvb, vT);
    attn_wmma<<<dim3(N_TOK / 16, NH), 32, 0, stream>>>(qb, kb16, vT, ob, N_TOK);
    run_gemm(ob, oW_l, h, N_TOK, DIM, NH * VH, 1, stream);

    rmsnorm_kernel<<<N_TOK, 256, 0, stream>>>(h, ffn_nw + (size_t)l * DIM, x,
                                              DIM, DIM, 0, DIM);
    run_gemm(x, w1_l, t1, N_TOK, INTER, DIM, 0, stream);
    run_gemm(x, w2_l, t2, N_TOK, INTER, DIM, 0, stream);
    silu_mul_kernel<<<(N_TOK * INTER) / 256, 256, 0, stream>>>(t1, t2,
                                                               N_TOK * INTER);
    run_gemm(t1, w3_l, h, N_TOK, DIM, INTER, 1, stream);
  }

  rmsnorm_kernel<<<N_TOK, 256, 0, stream>>>(h, norm_w, x, DIM, DIM, 0, DIM);
  run_gemm(x, outW, logits, N_TOK, VOCAB, DIM, 0, stream);
}
